// Encoder_16630113370599
// MI455X (gfx1250) — matrix operand order repaired and checked by reading, not compiled, not run on hardware
//
#include <hip/hip_runtime.h>
#include <stdint.h>

// ---------------------------------------------------------------------------
// Types for gfx1250 WMMA (wave32)
// ---------------------------------------------------------------------------
typedef __attribute__((ext_vector_type(16))) _Float16 v16h;
typedef __attribute__((ext_vector_type(8)))  float    v8f;
typedef unsigned int v4u __attribute__((ext_vector_type(4)));
typedef int          v8i __attribute__((ext_vector_type(8)));
typedef int          v4i __attribute__((ext_vector_type(4)));

union U16 {
    v16h     v;
    uint32_t u[8];
    _Float16 hh[16];
};

__device__ __forceinline__ v8f wmma_f16(const U16& a, const U16& b, v8f c) {
    // D = A(16x32 f16) * B(32x16 f16) + C(16x16 f32)
    return __builtin_amdgcn_wmma_f32_16x16x32_f16(
        false, a.v, false, b.v, (short)0, c, false, false);
}

__device__ __forceinline__ float gelu_tanh(float x) {
    float x3 = x * x * x;
    return 0.5f * x * (1.0f + tanhf(0.7978845608f * (x + 0.044715f * x3)));
}

// ---------------------------------------------------------------------------
// Problem constants
// ---------------------------------------------------------------------------
#define BB   4
#define NN   1024
#define DD   512
#define HH   8
#define DK   64
#define DHID 1024
#define MROW (BB * NN)          // 4096
#define QKVN (HH * 3 * DK)      // 1536

// ---------------------------------------------------------------------------
// Tensor Data Mover availability (device pass only)
// ---------------------------------------------------------------------------
#if defined(__AMDGCN__) && defined(__has_builtin)
#if __has_builtin(__builtin_amdgcn_tensor_load_to_lds) && \
    __has_builtin(__builtin_amdgcn_s_wait_tensorcnt)
#define HAVE_TDM 1
#endif
#endif
#ifndef HAVE_TDM
#define HAVE_TDM 0
#endif

#if HAVE_TDM
// Issue a TDM load of a [tile_d1 rows x tile_d0 elems] f16 tile from a 2-D
// tensor (tensor_d1 rows x tensor_d0 elems, row stride = stride0 elems).
// gsrc points at the tile start; lds_off is the destination LDS byte address.
__device__ __forceinline__ void tdm_load_2d(const _Float16* gsrc, uint32_t lds_off,
                                            int tensor_d0, int tensor_d1,
                                            int tile_d0, int tile_d1, int stride0) {
    uint64_t ga = (uint64_t)(uintptr_t)gsrc;
    v4u g0;
    g0[0] = 1u;                                  // count=1 (valid descriptor)
    g0[1] = lds_off;                             // LDS byte address [63:32]
    g0[2] = (uint32_t)ga;                        // global addr [95:64]
    g0[3] = (uint32_t)(ga >> 32) | (2u << 30);   // addr[56:32] | type=2 ("image")
    v8i g1;
    g1[0] = (int)(1u << 16);                                         // data_size=1 -> 2B
    g1[1] = (int)(((uint32_t)tensor_d0 & 0xFFFFu) << 16);            // tensor_dim0 lo16
    g1[2] = (int)((((uint32_t)tensor_d0 >> 16) & 0xFFFFu) |
                  (((uint32_t)tensor_d1 & 0xFFFFu) << 16));          // dim0 hi | dim1 lo
    g1[3] = (int)((((uint32_t)tensor_d1 >> 16) & 0xFFFFu) |
                  (((uint32_t)tile_d0 & 0xFFFFu) << 16));            // dim1 hi | tile_dim0
    g1[4] = tile_d1;                                                 // tile_dim1, tile_dim2=0
    g1[5] = stride0;                                                 // tensor_dim0_stride lo32
    g1[6] = 0;
    g1[7] = 0;
    v4i z4 = {0, 0, 0, 0};
#if defined(__clang_major__) && (__clang_major__ >= 23)
    v8i z8 = {0, 0, 0, 0, 0, 0, 0, 0};
    __builtin_amdgcn_tensor_load_to_lds(g0, g1, z4, z4, z8, 0);
#else
    __builtin_amdgcn_tensor_load_to_lds(g0, g1, z4, z4, 0);
#endif
}
#endif

// ---------------------------------------------------------------------------
// Shared GEMM core: block = 128 threads (4 waves), computes a 64(M) x 64(N)
// tile of A[M,Kd] * Bt[N,Kd]^T. Wave w owns rows [mbase+16w, +16); all waves
// share the B tile (64 rows x 32 k, f16) double-buffered in LDS, staged by
// the Tensor Data Mover (wave 0) and published with s_wait_tensorcnt+barrier.
// B fragments are preloaded into 4 register sets so the ds_load_b128 pairs
// pipeline ahead of the 4-WMMA chain. smem must hold 2*2048 f16 (8 KB).
// ---------------------------------------------------------------------------
__device__ __forceinline__ void gemm_core64(const _Float16* __restrict__ arow,
                                            const _Float16* __restrict__ Bt,
                                            int Nrows, int Kd, int nbase0,
                                            _Float16* smem,
                                            int wave, int half, int l15,
                                            v8f acc[4]) {
#if HAVE_TDM
    if (wave == 0)
        tdm_load_2d(Bt + (size_t)nbase0 * Kd, (uint32_t)(uintptr_t)smem,
                    Kd, Nrows, 32, 64, Kd);
#else
    {
        int t = threadIdx.x;
        const _Float16* src = Bt + (size_t)nbase0 * Kd;
#pragma unroll
        for (int j = 0; j < 2; ++j) {
            int idx = t + j * 128;          // 256 x 16B units = 4KB
            int r = idx >> 2;
            int c = (idx & 3) << 3;
            *(uint4*)(smem + r * 32 + c) = *(const uint4*)(src + (size_t)r * Kd + c);
        }
    }
#endif
    for (int k0 = 0; k0 < Kd; k0 += 32) {
        const int stage = (k0 >> 5) & 1;
        _Float16* sb = smem + stage * 2048;
#if HAVE_TDM
        if (wave == 0) __builtin_amdgcn_s_wait_tensorcnt(0);
#endif
        __syncthreads();
        if (k0 + 32 < Kd) {
#if HAVE_TDM
            if (wave == 0)
                tdm_load_2d(Bt + (size_t)nbase0 * Kd + (k0 + 32),
                            (uint32_t)(uintptr_t)(smem + (stage ^ 1) * 2048),
                            Kd, Nrows, 32, 64, Kd);
#else
            int t = threadIdx.x;
            const _Float16* src = Bt + (size_t)nbase0 * Kd + (k0 + 32);
            _Float16* dst = smem + (stage ^ 1) * 2048;
#pragma unroll
            for (int j = 0; j < 2; ++j) {
                int idx = t + j * 128;
                int r = idx >> 2;
                int c = (idx & 3) << 3;
                *(uint4*)(dst + r * 32 + c) = *(const uint4*)(src + (size_t)r * Kd + c);
            }
#endif
            __builtin_prefetch(arow + k0 + 32, 0, 3);   // next A chunk -> global_prefetch
        }
        U16 a;
#pragma unroll
        for (int v = 0; v < 8; ++v) {
            int ka = k0 + (half << 4) + (v << 1);
            a.u[v] = *(const uint32_t*)(const void*)(arow + ka);
        }
        U16 b4[4];
#pragma unroll
        for (int ct = 0; ct < 4; ++ct) {
            const uint32_t* bp =
                (const uint32_t*)(const void*)(sb + (ct * 16 + l15) * 32 + (half << 4));
#pragma unroll
            for (int v = 0; v < 8; ++v) b4[ct].u[v] = bp[v];   // 2x ds_load_b128
        }
#pragma unroll
        for (int ct = 0; ct < 4; ++ct)
            acc[ct] = wmma_f16(a, b4[ct], acc[ct]);
    }
}

// ---------------------------------------------------------------------------
// f32 -> f16 conversion
// ---------------------------------------------------------------------------
__global__ void cvt_f16(const float* __restrict__ src, _Float16* __restrict__ dst, int n) {
    int i = blockIdx.x * 256 + threadIdx.x;
    if (i < n) dst[i] = (_Float16)src[i];
}

// Build transposed+fused QKV weight [1536][512] (f16) and fused bias [1536] (f32).
// Row layout: row = (h*3 + which)*64 + k ; element = proj[h][d][k]
__global__ void prep_qkvw(const float* __restrict__ qw, const float* __restrict__ kw,
                          const float* __restrict__ vw, const float* __restrict__ qb,
                          const float* __restrict__ kb, const float* __restrict__ vb,
                          _Float16* __restrict__ wT, float* __restrict__ bias) {
    int i = blockIdx.x * 256 + threadIdx.x;     // [0, 1536*512)
    int row = i >> 9, d = i & 511;
    int h = row / 192, rem = row - h * 192;
    int wsel = rem >> 6, k = rem & 63;
    const float* src = (wsel == 0) ? qw : (wsel == 1) ? kw : vw;
    wT[i] = (_Float16)src[((h << 9) + d) * DK + k];
    if (i < QKVN) {
        int h2 = i / 192, rem2 = i - h2 * 192;
        int w2 = rem2 >> 6, k2 = rem2 & 63;
        const float* bs = (w2 == 0) ? qb : (w2 == 1) ? kb : vb;
        bias[i] = bs[(h2 << 6) + k2];
    }
}

// ---------------------------------------------------------------------------
// Generic WMMA GEMM: C[M,N] = act(A[M,Kd] * Bt[N,Kd]^T + bias[N])
// block = 128 (4 waves), grid = (M/64, N/64).
// ACT: 0 none, 1 gelu.  OUTF16: 0 -> f32 out, 1 -> f16 out.
// ---------------------------------------------------------------------------
template <int ACT, int OUTF16>
__global__ void gemm16(const _Float16* __restrict__ A, const _Float16* __restrict__ Bt,
                       const float* __restrict__ bias, void* __restrict__ Cout,
                       int M, int N, int Kd) {
    __shared__ _Float16 smem[2 * 2048];
    const int lane = threadIdx.x & 31;
    const int wave = threadIdx.x >> 5;
    const int half = lane >> 4;
    const int l15  = lane & 15;
    const int mbase  = blockIdx.x * 64 + wave * 16;
    const int nbase0 = blockIdx.y * 64;

    const _Float16* arow = A + (size_t)(mbase + l15) * Kd;
    v8f acc[4];
#pragma unroll
    for (int t = 0; t < 4; ++t) acc[t] = (v8f){0.f, 0.f, 0.f, 0.f, 0.f, 0.f, 0.f, 0.f};

    gemm_core64(arow, Bt, N, Kd, nbase0, smem, wave, half, l15, acc);

#pragma unroll
    for (int ct = 0; ct < 4; ++ct) {
        const int n = nbase0 + ct * 16 + l15;
        const float bv = bias ? bias[n] : 0.0f;
#pragma unroll
        for (int r = 0; r < 8; ++r) {
            int m = mbase + (half << 3) + r;
            float c = acc[ct][r] + bv;
            if (ACT == 1) c = gelu_tanh(c);
            if (OUTF16) ((_Float16*)Cout)[(size_t)m * N + n] = (_Float16)c;
            else        ((float*)Cout)[(size_t)m * N + n] = c;
        }
    }
}

// ---------------------------------------------------------------------------
// Fused QKV GEMM: A = xh[4096,512], Bt = wqkvT[1536,512], bias = qkvb[1536].
// Writes Qh[b,h,n,k], Kh[b,h,n,k] and Vt[b,h,k,n] (transposed V), all f16.
// block = 128, grid = (4096/64, 1536/64).
// ---------------------------------------------------------------------------
__global__ void qkv_gemm(const _Float16* __restrict__ A, const _Float16* __restrict__ Bt,
                         const float* __restrict__ bias,
                         _Float16* __restrict__ Qh, _Float16* __restrict__ Kh,
                         _Float16* __restrict__ Vt) {
    __shared__ _Float16 smem[2 * 2048];
    const int lane = threadIdx.x & 31;
    const int wave = threadIdx.x >> 5;
    const int half = lane >> 4;
    const int l15  = lane & 15;
    const int mbase  = blockIdx.x * 64 + wave * 16;
    const int nbase0 = blockIdx.y * 64;

    const _Float16* arow = A + (size_t)(mbase + l15) * DD;
    v8f acc[4];
#pragma unroll
    for (int t = 0; t < 4; ++t) acc[t] = (v8f){0.f, 0.f, 0.f, 0.f, 0.f, 0.f, 0.f, 0.f};

    gemm_core64(arow, Bt, QKVN, DD, nbase0, smem, wave, half, l15, acc);

#pragma unroll
    for (int ct = 0; ct < 4; ++ct) {
        const int col  = nbase0 + ct * 16 + l15;      // (h*3+wsel)*64 + kout
        const int h    = col / 192;
        const int rem  = col - h * 192;
        const int wsel = rem >> 6;                    // uniform per 16-col tile
        const int kout = rem & 63;
        const float bv = bias[col];
#pragma unroll
        for (int r = 0; r < 8; ++r) {
            int grow = mbase + (half << 3) + r;       // global row in [0,4096)
            int b_ = grow >> 10;
            int n  = grow & 1023;
            _Float16 hv = (_Float16)(acc[ct][r] + bv);
            if (wsel == 0)      Qh[(((size_t)(b_ * HH + h) * NN + n) << 6) + kout] = hv;
            else if (wsel == 1) Kh[(((size_t)(b_ * HH + h) * NN + n) << 6) + kout] = hv;
            else                Vt[((size_t)(b_ * HH + h) * DK + kout) * NN + n]   = hv;
        }
    }
}

// ---------------------------------------------------------------------------
// RBF-banded flash attention (transposed formulation) with TDM-staged K/V.
// Block = 128 threads (4 waves); all 4 waves share one (b,h) and walk the same
// key sequence, so the K chunk (32 keys x 64) and V^T chunk (64 vk x 32 keys)
// are staged once per block in double-buffered LDS by the Tensor Data Mover.
// Wave w owns query tile blockIdx.y*4+w. S^T = K*Q^T so each lane's C-column
// is a query; softmax reductions are in-lane + shfl_xor(16).
// grid = (B*H, 16), block = 128.
// ---------------------------------------------------------------------------
__global__ void attn_kernel(const _Float16* __restrict__ Qh, const _Float16* __restrict__ Kh,
                            const _Float16* __restrict__ Vt, const float* __restrict__ coords,
                            _Float16* __restrict__ obuf) {
    __shared__ _Float16 sK[2 * 2048];   // [stage][key32][kdim64]
    __shared__ _Float16 sV[2 * 2048];   // [stage][vk64][key32]
    const int lane = threadIdx.x & 31;
    const int wave = threadIdx.x >> 5;
    const int half = lane >> 4;
    const int l15  = lane & 15;
    const int bh = blockIdx.x;                 // 0..31
    const int b  = bh >> 3;
    const int h  = bh & 7;
    const int qbase = (blockIdx.y * 4 + wave) * 16;
    const int q = qbase + l15;

    const _Float16* Qb = Qh + ((size_t)bh * NN << 6);
    const _Float16* Kb = Kh + ((size_t)bh * NN << 6);
    const _Float16* Vb = Vt + ((size_t)bh * DK) * NN;
    const float*    cb = coords + (size_t)b * NN * 3;

    const float qx = cb[q * 3 + 0], qy = cb[q * 3 + 1], qz = cb[q * 3 + 2];

    // spread[h] = 1 + 5*(20^(h/7) - 1)/19 ;  log2(20) = 4.3219281
    const float sp = 1.0f + 5.0f * (exp2f((4.32192809f / 7.0f) * (float)h) - 1.0f) / 19.0f;
    const float s2 = sp * sp;
    const float coef = 1.0f / s2;                    // beta/(2*s^2), beta = 2
    const float lo = -2.0f * s2 * logf(0.99f);       // rbf <= 0.99
    const float hi = -2.0f * s2 * logf(0.01f);       // rbf >= 0.01

    // Q^T as WMMA B operand (lane column = query), two 32-wide kdim chunks.
    U16 qb0, qb1;
#pragma unroll
    for (int v = 0; v < 8; ++v) {
        int kq = ((v >> 2) << 4) + (half << 3) + ((v & 3) << 1);
        qb0.u[v] = *(const uint32_t*)(const void*)(Qb + ((size_t)q << 6) + kq);
        qb1.u[v] = *(const uint32_t*)(const void*)(Qb + ((size_t)q << 6) + 32 + kq);
    }

    // Stage the first K/V chunks.
#if HAVE_TDM
    if (wave == 0) {
        tdm_load_2d(Kb, (uint32_t)(uintptr_t)sK, 64, NN, 64, 32, 64);
        tdm_load_2d(Vb, (uint32_t)(uintptr_t)sV, NN, DK, 32, 64, NN);
    }
#else
    {
        int t = threadIdx.x;
#pragma unroll
        for (int j = 0; j < 2; ++j) {
            int idx = t + j * 128;               // K: 32 rows x 64 halves
            int r = idx >> 3, c = (idx & 7) << 3;
            *(uint4*)(sK + r * 64 + c) = *(const uint4*)(Kb + (size_t)r * 64 + c);
            int r2 = idx >> 2, c2 = (idx & 3) << 3;   // V: 64 rows x 32 halves
            *(uint4*)(sV + r2 * 32 + c2) = *(const uint4*)(Vb + (size_t)r2 * NN + c2);
        }
    }
#endif

    float mrow = -1e30f, lrow = 0.0f;
    v8f O[4];
#pragma unroll
    for (int t = 0; t < 4; ++t) O[t] = (v8f){0.f, 0.f, 0.f, 0.f, 0.f, 0.f, 0.f, 0.f};

    for (int kb = 0; kb < NN; kb += 32) {
        const int stage = (kb >> 5) & 1;
        const _Float16* sk = sK + stage * 2048;
        const _Float16* sv = sV + stage * 2048;
#if HAVE_TDM
        if (wave == 0) __builtin_amdgcn_s_wait_tensorcnt(0);
#endif
        __syncthreads();
        if (kb + 32 < NN) {
#if HAVE_TDM
            if (wave == 0) {
                tdm_load_2d(Kb + (size_t)(kb + 32) * 64,
                            (uint32_t)(uintptr_t)(sK + (stage ^ 1) * 2048),
                            64, NN, 64, 32, 64);
                tdm_load_2d(Vb + (kb + 32),
                            (uint32_t)(uintptr_t)(sV + (stage ^ 1) * 2048),
                            NN, DK, 32, 64, NN);
            }
#else
            int t = threadIdx.x;
            _Float16* dK = sK + (stage ^ 1) * 2048;
            _Float16* dV = sV + (stage ^ 1) * 2048;
#pragma unroll
            for (int j = 0; j < 2; ++j) {
                int idx = t + j * 128;
                int r = idx >> 3, c = (idx & 7) << 3;
                *(uint4*)(dK + r * 64 + c) =
                    *(const uint4*)(Kb + (size_t)(kb + 32 + r) * 64 + c);
                int r2 = idx >> 2, c2 = (idx & 3) << 3;
                *(uint4*)(dV + r2 * 32 + c2) =
                    *(const uint4*)(Vb + (size_t)r2 * NN + kb + 32 + c2);
            }
#endif
        }

        // --- S^T for two 16-key tiles (A fragments preloaded from LDS) ---
        U16 ak[4];
#pragma unroll
        for (int t = 0; t < 2; ++t) {
            const _Float16* krow = sk + (size_t)(t * 16 + l15) * 64;
#pragma unroll
            for (int v = 0; v < 8; ++v) {
                int ka = ((v >> 2) << 4) + (half << 3) + ((v & 3) << 1);
                ak[2 * t + 0].u[v] = *(const uint32_t*)(const void*)(krow + ka);
                ak[2 * t + 1].u[v] = *(const uint32_t*)(const void*)(krow + 32 + ka);
            }
        }
        v8f st[2];
#pragma unroll
        for (int t = 0; t < 2; ++t) {
            v8f s = {0.f, 0.f, 0.f, 0.f, 0.f, 0.f, 0.f, 0.f};
            s = wmma_f16(ak[2 * t + 0], qb0, s);
            s = wmma_f16(ak[2 * t + 1], qb1, s);
            st[t] = s;
        }

        // --- per-lane key coords, broadcast by shuffle ---
        const float kx = cb[(kb + lane) * 3 + 0];
        const float ky = cb[(kb + lane) * 3 + 1];
        const float kz = cb[(kb + lane) * 3 + 2];

        float lt[16];
        unsigned vm = 0u;
#pragma unroll
        for (int t = 0; t < 2; ++t) {
#pragma unroll
            for (int r = 0; r < 8; ++r) {
                int idx = t * 16 + (half << 3) + r;          // key lane within block
                float dx = qx - __shfl(kx, idx, 32);
                float dy = qy - __shfl(ky, idx, 32);
                float dz = qz - __shfl(kz, idx, 32);
                float d2 = dx * dx + dy * dy + dz * dz;
                bool ok = (d2 >= lo) && (d2 <= hi);
                float lg = 0.125f * st[t][r] - coef * d2;    // S/sqrt(64) + beta*log(rbf)
                int i = t * 8 + r;
                lt[i] = ok ? lg : -1e30f;
                vm |= (ok ? 1u : 0u) << i;
            }
        }

        // --- online softmax (per query column) ---
        float bm = lt[0];
#pragma unroll
        for (int i = 1; i < 16; ++i) bm = fmaxf(bm, lt[i]);
        bm = fmaxf(bm, __shfl_xor(bm, 16, 32));
        float mnew = fmaxf(mrow, bm);
        float scale = __expf(mrow - mnew);

        float p[16];
        float ps = 0.0f;
#pragma unroll
        for (int i = 0; i < 16; ++i) {
            p[i] = ((vm >> i) & 1u) ? __expf(lt[i] - mnew) : 0.0f;
            ps += p[i];
        }
        ps += __shfl_xor(ps, 16, 32);
        lrow = lrow * scale + ps;
        mrow = mnew;
#pragma unroll
        for (int t = 0; t < 4; ++t)
#pragma unroll
            for (int r = 0; r < 8; ++r) O[t][r] *= scale;

        // --- repack p^T into B-operand layout (half-wave swap + f16 pack) ---
        float pp[16];
#pragma unroll
        for (int i = 0; i < 16; ++i) pp[i] = __shfl_xor(p[i], 16, 32);

        U16 bm16;
#pragma unroll
        for (int v = 0; v < 4; ++v) {
            float e0 = half ? pp[8 + 2 * v]     : p[2 * v];
            float e1 = half ? pp[8 + 2 * v + 1] : p[2 * v + 1];
            float f0 = half ? p[8 + 2 * v]      : pp[2 * v];
            float f1 = half ? p[8 + 2 * v + 1]  : pp[2 * v + 1];
            bm16.hh[2 * v]         = (_Float16)e0;
            bm16.hh[2 * v + 1]     = (_Float16)e1;
            bm16.hh[8 + 2 * v]     = (_Float16)f0;
            bm16.hh[8 + 2 * v + 1] = (_Float16)f1;
        }

        // --- O^T += V^T * p^T, four 16-row vk tiles (A preloaded from LDS) ---
        U16 av[4];
#pragma unroll
        for (int vt = 0; vt < 4; ++vt) {
            const _Float16* vrow = sv + (size_t)(vt * 16 + l15) * 32;
#pragma unroll
            for (int v = 0; v < 8; ++v) {
                int ka = (half << 4) + (v << 1);
                av[vt].u[v] = *(const uint32_t*)(const void*)(vrow + ka);
            }
        }
#pragma unroll
        for (int vt = 0; vt < 4; ++vt)
            O[vt] = wmma_f16(av[vt], bm16, O[vt]);
    }

    // --- normalize and scatter into interleaved (k*H + h) layout ---
    const float inv = (lrow > 0.0f) ? (1.0f / lrow) : 0.0f;
    _Float16* orow = obuf + ((size_t)(b * NN + q) << 9);   // *512
#pragma unroll
    for (int vt = 0; vt < 4; ++vt) {
#pragma unroll
        for (int r = 0; r < 8; ++r) {
            int vk = vt * 16 + (half << 3) + r;
            orow[(vk << 3) + h] = (_Float16)(O[vt][r] * inv);
        }
    }
}

// ---------------------------------------------------------------------------
// LayerNorm with fused residual: y = LN(xres + delta)*g + b.
// Writes f32 (always) and optional f16 copy. block=256, grid=4096.
// ---------------------------------------------------------------------------
__global__ void ln_kernel(const float* __restrict__ xres, const float* __restrict__ delta,
                          const float* __restrict__ g, const float* __restrict__ bta,
                          float* __restrict__ yout, _Float16* __restrict__ yh) {
    __shared__ float red[8];
    const int row = blockIdx.x;
    const int tid = threadIdx.x;
    const float* xr = xres + (size_t)row * DD;
    const float* dr = delta + (size_t)row * DD;

    float v0 = xr[tid] + dr[tid];
    float v1 = xr[tid + 256] + dr[tid + 256];

    float s = v0 + v1;
#pragma unroll
    for (int o = 16; o > 0; o >>= 1) s += __shfl_xor(s, o, 32);
    if ((tid & 31) == 0) red[tid >> 5] = s;
    __syncthreads();
    if (tid == 0) {
        float t = 0.f;
#pragma unroll
        for (int i = 0; i < 8; ++i) t += red[i];
        red[0] = t;
    }
    __syncthreads();
    const float mean = red[0] * (1.0f / DD);
    __syncthreads();

    float d0 = v0 - mean, d1 = v1 - mean;
    float s2 = d0 * d0 + d1 * d1;
#pragma unroll
    for (int o = 16; o > 0; o >>= 1) s2 += __shfl_xor(s2, o, 32);
    if ((tid & 31) == 0) red[tid >> 5] = s2;
    __syncthreads();
    if (tid == 0) {
        float t = 0.f;
#pragma unroll
        for (int i = 0; i < 8; ++i) t += red[i];
        red[0] = t;
    }
    __syncthreads();
    const float var = red[0] * (1.0f / DD);
    const float inv = rsqrtf(var + 1e-5f);

    float o0 = d0 * inv * g[tid] + bta[tid];
    float o1 = d1 * inv * g[tid + 256] + bta[tid + 256];
    yout[(size_t)row * DD + tid] = o0;
    yout[(size_t)row * DD + tid + 256] = o1;
    if (yh) {
        yh[(size_t)row * DD + tid] = (_Float16)o0;
        yh[(size_t)row * DD + tid + 256] = (_Float16)o1;
    }
}

// ---------------------------------------------------------------------------
// Launch
// ---------------------------------------------------------------------------
extern "C" void kernel_launch(void* const* d_in, const int* in_sizes, int n_in,
                              void* d_out, int out_size, void* d_ws, size_t ws_size,
                              hipStream_t stream) {
    (void)in_sizes; (void)n_in; (void)out_size; (void)ws_size;

    const float* x      = (const float*)d_in[0];
    const float* coords = (const float*)d_in[1];
    const float* q_proj = (const float*)d_in[2];
    const float* k_proj = (const float*)d_in[3];
    const float* v_proj = (const float*)d_in[4];
    const float* q_bias = (const float*)d_in[5];
    const float* k_bias = (const float*)d_in[6];
    const float* v_bias = (const float*)d_in[7];
    const float* w_out  = (const float*)d_in[8];
    const float* ln1_g  = (const float*)d_in[9];
    const float* ln1_b  = (const float*)d_in[10];
    const float* w1     = (const float*)d_in[11];
    const float* b1     = (const float*)d_in[12];
    const float* w2     = (const float*)d_in[13];
    const float* b2     = (const float*)d_in[14];
    const float* ln2_g  = (const float*)d_in[15];
    const float* ln2_b  = (const float*)d_in[16];

    char* w = (char*)d_ws;
    size_t off = 0;
    auto carve = [&](size_t bytes) -> void* {
        void* p = w + off;
        off = (off + bytes + 255) & ~(size_t)255;
        return p;
    };

    _Float16* xh    = (_Float16*)carve((size_t)MROW * DD * 2);
    _Float16* wqkvT = (_Float16*)carve((size_t)QKVN * DD * 2);
    float*    qkvb  = (float*)   carve((size_t)QKVN * 4);
    _Float16* wouth = (_Float16*)carve((size_t)DD * DD * 2);
    _Float16* w1h   = (_Float16*)carve((size_t)DHID * DD * 2);
    _Float16* w2h   = (_Float16*)carve((size_t)DD * DHID * 2);
    _Float16* Qh    = (_Float16*)carve((size_t)BB * HH * NN * DK * 2);
    _Float16* Kh    = (_Float16*)carve((size_t)BB * HH * NN * DK * 2);
    _Float16* Vt    = (_Float16*)carve((size_t)BB * HH * DK * NN * 2);
    _Float16* obuf  = (_Float16*)carve((size_t)MROW * DD * 2);
    float*    attnf = (float*)   carve((size_t)MROW * DD * 4);
    float*    ybuf  = (float*)   carve((size_t)MROW * DD * 4);
    _Float16* yh    = (_Float16*)carve((size_t)MROW * DD * 2);
    _Float16* hbuf  = (_Float16*)carve((size_t)MROW * DHID * 2);
    float*    ffnf  = (float*)   carve((size_t)MROW * DD * 4);

    // 1) f16 conversions
    cvt_f16<<<(MROW * DD) / 256, 256, 0, stream>>>(x, xh, MROW * DD);
    cvt_f16<<<(DD * DD) / 256, 256, 0, stream>>>(w_out, wouth, DD * DD);
    cvt_f16<<<(DHID * DD) / 256, 256, 0, stream>>>(w1, w1h, DHID * DD);
    cvt_f16<<<(DD * DHID) / 256, 256, 0, stream>>>(w2, w2h, DD * DHID);
    prep_qkvw<<<(QKVN * DD) / 256, 256, 0, stream>>>(q_proj, k_proj, v_proj,
                                                     q_bias, k_bias, v_bias, wqkvT, qkvb);

    // 2) fused QKV projection (writes Q, K row-major; V transposed)
    qkv_gemm<<<dim3(MROW / 64, QKVN / 64), 128, 0, stream>>>(xh, wqkvT, qkvb, Qh, Kh, Vt);

    // 3) RBF-banded flash attention (TDM-staged K/V)
    attn_kernel<<<dim3(BB * HH, 16), 128, 0, stream>>>(Qh, Kh, Vt, coords, obuf);

    // 4) output projection (no bias)
    gemm16<0, 0><<<dim3(MROW / 64, DD / 64), 128, 0, stream>>>(obuf, wouth, nullptr,
                                                               attnf, MROW, DD, DD);
    // 5) residual + LN1 (f32 + f16 outputs)
    ln_kernel<<<MROW, 256, 0, stream>>>(x, attnf, ln1_g, ln1_b, ybuf, yh);

    // 6) FFN1 with fused GELU -> f16
    gemm16<1, 1><<<dim3(MROW / 64, DHID / 64), 128, 0, stream>>>(yh, w1h, b1,
                                                                 hbuf, MROW, DHID, DD);
    // 7) FFN2 (+b2) -> f32
    gemm16<0, 0><<<dim3(MROW / 64, DD / 64), 128, 0, stream>>>(hbuf, w2h, b2,
                                                               ffnf, MROW, DD, DHID);
    // 8) residual + LN2 -> d_out
    ln_kernel<<<MROW, 256, 0, stream>>>(ybuf, ffnf, ln2_g, ln2_b, (float*)d_out, nullptr);
}
